// RNN_51943334478543
// MI455X (gfx1250) — matrix operand order repaired and checked by reading, not compiled, not run on hardware
//
#include <hip/hip_runtime.h>
#include <hip/hip_bf16.h>
#include <math.h>

// Problem constants (match reference)
#define T_SEQ 512
#define BATCH 64
#define ISZ   256
#define HID   512
#define BH    (BATCH * HID)
#define NBLK2 8   // persistent workgroups in the recurrence kernel

typedef __attribute__((ext_vector_type(16))) __bf16         v16bf;
typedef __attribute__((ext_vector_type(8)))  float          v8f;
typedef __attribute__((ext_vector_type(16))) unsigned short v16us;
typedef __attribute__((ext_vector_type(4)))  unsigned int   u32x4;
typedef __attribute__((ext_vector_type(8)))  unsigned int   u32x8;

union Frag {
    v16bf b;
    v16us u;
    uint4 q[2];
};

// float -> bf16 bits, round-to-nearest-even (scalar fallback)
__device__ __forceinline__ unsigned short f2bf(float f) {
    unsigned u = __float_as_uint(f);
    u += 0x7FFFu + ((u >> 16) & 1u);
    return (unsigned short)(u >> 16);
}

// pack two floats into {lo=a, hi=b} bf16 pair; prefer v_cvt_pk_bf16_f32
#if defined(__has_builtin)
#if __has_builtin(__builtin_amdgcn_cvt_pk_bf16_f32)
#define HAVE_CVT_PK_BF16 1
#endif
#endif

__device__ __forceinline__ unsigned pk2bf(float a, float b) {
#ifdef HAVE_CVT_PK_BF16
    typedef __attribute__((ext_vector_type(2))) __bf16 v2bf;
    union { v2bf v; unsigned u; } r;
    r.v = __builtin_amdgcn_cvt_pk_bf16_f32(a, b);
    return r.u;
#else
    return (unsigned)f2bf(a) | ((unsigned)f2bf(b) << 16);
#endif
}

// ---------------------------------------------------------------------------
// Pack W (rows x cols fp32, row-major) into the per-lane bf16 B-operand layout
// of v_wmma_f32_16x16x32_bf16:  B[k][n] = W[n][k].
// Fragment (nt over rows/16, kt over cols/32), lane l:
//   n = nt*16 + (l&15), kbase = kt*32 + (l<16 ? 0 : 16)
//   dst[((nt*KT + kt)*32 + l)*16 + j] = bf16(W[n][kbase + j]),  j = 0..15
// ---------------------------------------------------------------------------
__global__ void rnn_pack_b_kernel(const float* __restrict__ W,
                                  unsigned short* __restrict__ dst,
                                  int rows, int cols) {
    int KT = cols >> 5;
    int total = (rows >> 4) * KT * 32;
    int g = blockIdx.x * blockDim.x + threadIdx.x;
    if (g >= total) return;
    int l  = g & 31;
    int kt = (g >> 5) % KT;
    int nt = (g >> 5) / KT;
    int n  = nt * 16 + (l & 15);
    int kb = kt * 32 + ((l < 16) ? 0 : 16);
    const float* src = W + (size_t)n * cols + kb;
    uint4 q0, q1;
    q0.x = pk2bf(src[0],  src[1]);  q0.y = pk2bf(src[2],  src[3]);
    q0.z = pk2bf(src[4],  src[5]);  q0.w = pk2bf(src[6],  src[7]);
    q1.x = pk2bf(src[8],  src[9]);  q1.y = pk2bf(src[10], src[11]);
    q1.z = pk2bf(src[12], src[13]); q1.w = pk2bf(src[14], src[15]);
    uint4* d = (uint4*)(dst + (size_t)g * 16);
    d[0] = q0;
    d[1] = q1;
}

// h0 (fp32, B*H) -> bf16 mirror used by the TDM staging in step 0
__global__ void rnn_h0_to_bf_kernel(const float* __restrict__ h0,
                                    unsigned short* __restrict__ hbf) {
    int idx = blockIdx.x * blockDim.x + threadIdx.x;   // over BH/8
    if (idx >= BH / 8) return;
    const float4* s = ((const float4*)h0) + idx * 2;
    float4 f0 = s[0], f1 = s[1];
    uint4 q;
    q.x = pk2bf(f0.x, f0.y); q.y = pk2bf(f0.z, f0.w);
    q.z = pk2bf(f1.x, f1.y); q.w = pk2bf(f1.z, f1.w);
    ((uint4*)hbf)[idx] = q;
}

// ---------------------------------------------------------------------------
// Phase 1: xW = x @ W_ih^T + b_ih + b_hh for all timesteps (one big GEMM).
// M = T*B = 32768 rows, K = 256, N = 512. Block = 128 threads (4 waves),
// wave handles a 16(M) x 64(N) strip = 4 fp32 accumulators, K-chained WMMA.
// Writes pre-activation into d_out[t*B*H ...] (recurrence rewrites in place).
// ---------------------------------------------------------------------------
__global__ __launch_bounds__(128)
void rnn_xproj_kernel(const float* __restrict__ x,
                      const unsigned short* __restrict__ wih_pk,
                      const float* __restrict__ b_ih,
                      const float* __restrict__ b_hh,
                      float* __restrict__ out) {
    const int KT = ISZ / 32;                 // 8
    int lane  = threadIdx.x & 31;
    int wave  = threadIdx.x >> 5;            // 0..3
    int Mbase = blockIdx.x * 64 + wave * 16; // row tile base in T*B
    int Nblk  = blockIdx.y * 64;             // col tile base in H
    int lm    = lane & 15;
    int koff  = (lane >> 4) << 4;            // 0 or 16 per A-fragment layout

    v8f acc[4] = {{}, {}, {}, {}};

    const float* xrow = x + (size_t)(Mbase + lm) * ISZ;

    for (int kt = 0; kt < KT; ++kt) {
        // A fragment: row M = lane&15, K = 16*(lane>=16) + s (contiguous order, same as packed B)
        Frag a;
        const float4* p0 = (const float4*)(xrow + kt * 32 + koff);
        const float4* p1 = (const float4*)(xrow + kt * 32 + 8 + koff);
        float4 f0 = p0[0], f1 = p0[1], f2 = p1[0], f3 = p1[1];
        a.q[0].x = pk2bf(f0.x, f0.y); a.q[0].y = pk2bf(f0.z, f0.w);
        a.q[0].z = pk2bf(f1.x, f1.y); a.q[0].w = pk2bf(f1.z, f1.w);
        a.q[1].x = pk2bf(f2.x, f2.y); a.q[1].y = pk2bf(f2.z, f2.w);
        a.q[1].z = pk2bf(f3.x, f3.y); a.q[1].w = pk2bf(f3.z, f3.w);
#pragma unroll
        for (int nt = 0; nt < 4; ++nt) {
            int ntg = (Nblk >> 4) + nt;
            const v16bf* bp = (const v16bf*)(wih_pk +
                (((size_t)ntg * KT + kt) * 32 + lane) * 16);
            acc[nt] = __builtin_amdgcn_wmma_f32_16x16x32_bf16(
                false, a.b, false, *bp, (short)0, acc[nt], false, false);
        }
    }

#pragma unroll
    for (int nt = 0; nt < 4; ++nt) {
        int N = Nblk + nt * 16 + lm;
        float bias = b_ih[N] + b_hh[N];
#pragma unroll
        for (int v = 0; v < 8; ++v) {
            int row = Mbase + v + ((lane >> 4) << 3);   // C/D layout: M = v + 8*(lane>=16)
            out[(size_t)row * HID + N] = acc[nt][v] + bias;
        }
    }
}

// ---------------------------------------------------------------------------
// Phase 2: persistent sequential scan. 8 workgroups x 256 threads; block b
// owns H-slice [b*64, b*64+64). Per step:
//   - TDM (tensor_load_to_lds, TENSORcnt) stages the 64 KB bf16 mirror of
//     h_{t-1} from global into LDS — one DMA instruction, no VALU converts
//   - 16 k-tiles x 2 accumulators of v_wmma_f32_16x16x32_bf16 (A from LDS
//     via ds_load_b128, B from L2-resident packed W_hh)
//   - h_t = tanh(acc + xw_t): fp32 in place over d_out[t] + bf16 mirror
//   - device-wide step barrier: release atomicAdd + acquire spin on L2 counter
// ---------------------------------------------------------------------------
__global__ __launch_bounds__(256)
void rnn_scan_kernel(const unsigned short* __restrict__ whh_pk,
                     unsigned short* __restrict__ hbf,
                     float* __restrict__ out,
                     unsigned int* __restrict__ sync) {
    __shared__ unsigned short h_lds[BATCH * HID];    // 64 KB bf16 of h_{t-1}
    const int KT = HID / 32;                         // 16
    int tid  = threadIdx.x;
    int lane = tid & 31;
    int wave = tid >> 5;        // 0..7
    int mt   = wave & 3;        // M tile (batch/16)
    int nh   = wave >> 2;       // which pair of N tiles
    int lm   = lane & 15;
    int koff = (lane >> 4) << 4;
    int Nblk = blockIdx.x * 64;
    int nt0  = nh * 2, nt1 = nh * 2 + 1;
    size_t bbase0 = ((size_t)((Nblk >> 4) + nt0) * KT) * 32 * 16;
    size_t bbase1 = ((size_t)((Nblk >> 4) + nt1) * KT) * 32 * 16;

    // ---- Tensor DMA descriptor (loop-invariant): 1-D copy of 8192 x 8B
    // (= 64 KB) from the global bf16 h mirror into LDS offset 0.
    unsigned long long ga = (unsigned long long)(uintptr_t)hbf;
    u32x4 g0;
    g0[0] = 1u;                                   // count=1, user descriptor
    g0[1] = 0u;                                   // lds_addr: h_lds is the sole LDS alloc -> offset 0
    g0[2] = (unsigned)ga;                         // global_addr[31:0]
    g0[3] = (unsigned)((ga >> 32) & 0x01FFFFFFu)  // global_addr[56:32]
            | 0x80000000u;                        // type=2 ("image")
    u32x8 g1;
    g1[0] = 3u << 16;                             // wg_mask=0, data_size=8B
    g1[1] = (8192u & 0xFFFFu) << 16;              // tensor_dim0 = 8192 (lo16)
    g1[2] = 1u << 16;                             // tensor_dim0 hi16=0, tensor_dim1=1
    g1[3] = 8192u << 16;                          // tensor_dim1 hi=0, tile_dim0=8192
    g1[4] = 1u;                                   // tile_dim1=1, tile_dim2=0
    g1[5] = 8192u;                                // tensor_dim0_stride lo32
    g1[6] = 0u;                                   // stride hi, dim1_stride lo
    g1[7] = 0u;                                   // dim1_stride hi

    unsigned short* ldsp = &h_lds[0];   // escapes LDS object into the asm below

    for (int t = 0; t < T_SEQ; ++t) {
        // stage h_{t-1} (bf16 mirror) into LDS via the Tensor Data Mover.
        // %2 (LDS base pointer) is unused in the template but makes the LDS
        // object escape, so the "memory" clobber aliases h_lds and the
        // compiler cannot fold the ds_load_b128 fragment reads below.
        if (wave == 0) {
            asm volatile("tensor_load_to_lds %0, %1"
                         :: "s"(g0), "s"(g1), "v"(ldsp)
                         : "memory");
            __builtin_amdgcn_s_wait_tensorcnt(0);
        }
        __syncthreads();

        float* outp = out + (size_t)t * BH;
        // warm L2/L0 for the epilogue read-modify-write of xw
        __builtin_prefetch(outp + (size_t)(mt * 16 + lm) * HID + Nblk + lm, 0, 0);

        v8f acc0 = {}, acc1 = {};
#pragma unroll 4
        for (int kt = 0; kt < KT; ++kt) {
            Frag a;
            int arow = (mt * 16 + lm) * HID + kt * 32 + koff;
            a.q[0] = *(const uint4*)&h_lds[arow];
            a.q[1] = *(const uint4*)&h_lds[arow + 8];
            const v16bf* b0 = (const v16bf*)(whh_pk + bbase0 + ((size_t)kt * 32 + lane) * 16);
            const v16bf* b1 = (const v16bf*)(whh_pk + bbase1 + ((size_t)kt * 32 + lane) * 16);
            acc0 = __builtin_amdgcn_wmma_f32_16x16x32_bf16(
                false, a.b, false, *b0, (short)0, acc0, false, false);
            acc1 = __builtin_amdgcn_wmma_f32_16x16x32_bf16(
                false, a.b, false, *b1, (short)0, acc1, false, false);
        }

        // epilogue: h_t = tanh(xw_t + h_{t-1} @ W_hh^T); fp32 out + bf16 mirror
        int N0 = Nblk + nt0 * 16 + lm;
        int N1 = Nblk + nt1 * 16 + lm;
#pragma unroll
        for (int v = 0; v < 8; ++v) {
            int m = mt * 16 + v + ((lane >> 4) << 3);
            size_t i0 = (size_t)m * HID + N0;
            size_t i1 = (size_t)m * HID + N1;
            float hv0 = tanhf(acc0[v] + outp[i0]);
            float hv1 = tanhf(acc1[v] + outp[i1]);
            outp[i0] = hv0;
            outp[i1] = hv1;
            hbf[i0] = f2bf(hv0);
            hbf[i1] = f2bf(hv1);
            if (t == T_SEQ - 1) {           // h_last copy
                out[(size_t)T_SEQ * BH + i0] = hv0;
                out[(size_t)T_SEQ * BH + i1] = hv1;
            }
        }

        // device-wide step barrier (8 resident workgroups)
        __threadfence();
        __syncthreads();
        if (tid == 0)
            __hip_atomic_fetch_add(sync, 1u, __ATOMIC_RELEASE, __HIP_MEMORY_SCOPE_AGENT);
        unsigned target = (unsigned)NBLK2 * (unsigned)(t + 1);
        while (__hip_atomic_load(sync, __ATOMIC_ACQUIRE, __HIP_MEMORY_SCOPE_AGENT) < target)
            __builtin_amdgcn_s_sleep(1);
    }
}

extern "C" void kernel_launch(void* const* d_in, const int* in_sizes, int n_in,
                              void* d_out, int out_size, void* d_ws, size_t ws_size,
                              hipStream_t stream) {
    (void)in_sizes; (void)n_in; (void)out_size; (void)ws_size;
    const float* x   = (const float*)d_in[0];
    const float* h0  = (const float*)d_in[1];
    const float* Wih = (const float*)d_in[2];
    const float* Whh = (const float*)d_in[3];
    const float* bih = (const float*)d_in[4];
    const float* bhh = (const float*)d_in[5];
    float* out = (float*)d_out;

    unsigned char* ws = (unsigned char*)d_ws;
    unsigned int*   sync   = (unsigned int*)ws;                        // 256 B
    unsigned short* wih_pk = (unsigned short*)(ws + 512);              // 256 KB
    const size_t wih_bytes = (size_t)(HID / 16) * (ISZ / 32) * 32 * 16 * 2;
    unsigned short* whh_pk = (unsigned short*)(ws + 512 + wih_bytes);  // 512 KB
    const size_t whh_bytes = (size_t)(HID / 16) * (HID / 32) * 32 * 16 * 2;
    unsigned short* hbf    = (unsigned short*)(ws + 512 + wih_bytes + whh_bytes); // 64 KB

    // zero the step-barrier counter every call (graph-capture safe, deterministic)
    hipMemsetAsync(d_ws, 0, 256, stream);

    int e1 = (HID / 16) * (ISZ / 32) * 32;   // 8192 fragment-lane entries
    rnn_pack_b_kernel<<<(e1 + 255) / 256, 256, 0, stream>>>(Wih, wih_pk, HID, ISZ);
    int e2 = (HID / 16) * (HID / 32) * 32;   // 16384
    rnn_pack_b_kernel<<<(e2 + 255) / 256, 256, 0, stream>>>(Whh, whh_pk, HID, HID);
    rnn_h0_to_bf_kernel<<<(BH / 8 + 255) / 256, 256, 0, stream>>>(h0, hbf);

    dim3 g1(T_SEQ * BATCH / 64, HID / 64);   // (512, 8)
    rnn_xproj_kernel<<<g1, 128, 0, stream>>>(x, wih_pk, bih, bhh, out);

    rnn_scan_kernel<<<NBLK2, 256, 0, stream>>>(whh_pk, hbf, out, sync);
}
